// RecursiveSimplePtreeLayer_16192026706181
// MI455X (gfx1250) — matrix operand order repaired and checked by reading, not compiled, not run on hardware
//
#include <hip/hip_runtime.h>
#include <hip/hip_bf16.h>

// Sizes from the reference
#define HID   128
#define LSEQ  256
#define KCONV 4
#define BB    64
#define CC    16

#define DSTR  136          // padded LDS row stride (bf16 elems) -> 272B, spreads banks
#define NWAVE 8
#define TPB   (NWAVE * 32) // 256 threads = 8 wave32, each wave owns 2 M-tiles
#define MT    2            // M-tiles per wave (16 waves' worth of rows total)

typedef __attribute__((ext_vector_type(16))) __bf16 v16bf;
typedef __attribute__((ext_vector_type(8)))  float  v8f;

// async-to-LDS builtin operand types (from hipcc diagnostic: vector_size(16) int,
// AS1 = __device__, AS3 = __shared__)
typedef int v4i_t __attribute__((vector_size(16)));
typedef __attribute__((address_space(1))) v4i_t* as1_v4i;
typedef __attribute__((address_space(3))) v4i_t* as3_v4i;

union ABReg { v16bf v; uint4 q[2]; };

#if defined(__has_builtin)
#if __has_builtin(__builtin_amdgcn_global_load_async_to_lds_b128)
#define HAVE_ASYNC_LDS 1
#endif
#if __has_builtin(__builtin_amdgcn_s_wait_asynccnt)
#define HAVE_WAIT_ASYNC 1
#endif
#endif

__device__ __forceinline__ unsigned short f2bf(float f) {
  unsigned u = __float_as_uint(f);
  u += 0x7FFFu + ((u >> 16) & 1u);   // round-to-nearest-even truncation
  return (unsigned short)(u >> 16);
}

__device__ __forceinline__ float eluf(float x) {
  return x > 0.f ? x : expm1f(x);
}

// copy one 16B chunk global -> LDS (async-to-LDS when available)
__device__ __forceinline__ void stage16(const unsigned short* __restrict__ gsrc,
                                        unsigned short* ldst) {
#ifdef HAVE_ASYNC_LDS
  __builtin_amdgcn_global_load_async_to_lds_b128(
      (as1_v4i)gsrc, (as3_v4i)ldst, 0, 0);
#else
  *(uint4*)ldst = *(const uint4*)gsrc;
#endif
}

__device__ __forceinline__ void stage_fence() {
#ifdef HAVE_ASYNC_LDS
#ifdef HAVE_WAIT_ASYNC
  __builtin_amdgcn_s_wait_asynccnt(0);
#else
  asm volatile("s_wait_asynccnt 0x0" ::: "memory");
#endif
#endif
}

__global__ void cvt_bf16_kernel(const float* __restrict__ src,
                                unsigned short* __restrict__ dst, int n) {
  int i = blockIdx.x * blockDim.x + threadIdx.x;
  if (i < n) dst[i] = f2bf(src[i]);
}

// One workgroup per (b,c):
// z2[b,c,:] = elu(W_z_final @ sum_l elu(sum_k W_z[k] @ x[leaf[(l+k)%256]]))
__global__ __launch_bounds__(TPB) void ptree_z_kernel(
    const unsigned short* __restrict__ xbf,     // [N_NODES,128] bf16
    const int*            __restrict__ leaf_idx,// [B,C,L]
    const unsigned short* __restrict__ wzbf,    // [4,128,128] bf16
    const float*          __restrict__ wz_final,// [128,128] f32
    float*                __restrict__ z2out)   // [B*C,128] f32
{
  extern __shared__ char smem_raw[];
  unsigned short* dataL = (unsigned short*)smem_raw;          // LSEQ x DSTR bf16
  unsigned short* wL    = dataL + LSEQ * DSTR;                // 4*HID x DSTR bf16
  float*          zbarL = (float*)(wL + KCONV * HID * DSTR);  // HID f32

  const int bc   = blockIdx.x;
  const int tid  = threadIdx.x;
  const int lane = tid & 31;
  const int wv   = tid >> 5;       // wave id; owns M-tiles wv and wv+8
  const int m    = lane & 15;      // A: row within tile ; B: output column within tile
  const int hi   = lane >> 4;      // lane-half selects K sub-range

  // ---- stage gathered leaf rows into LDS (bf16, 16B chunks) ----
  const int* li = leaf_idx + bc * LSEQ;
  for (int i = tid; i < LSEQ * 16; i += TPB) {
    int r = i >> 4, ch = i & 15;
    stage16(xbf + (size_t)li[r] * HID + ch * 8, dataL + r * DSTR + ch * 8);
  }
  // ---- stage W_z (bf16) into LDS ----
  for (int i = tid; i < KCONV * HID * 16; i += TPB) {
    int r = i >> 4, ch = i & 15;
    stage16(wzbf + (size_t)r * HID + ch * 8, wL + r * DSTR + ch * 8);
  }
  if (tid < HID) zbarL[tid] = 0.f;
  stage_fence();
  __syncthreads();

  // ---- main GEMM: per wave, 2 M-tiles x 8 N-tiles; B loaded once, used twice ----
  v8f acc[MT][8] = {};

#pragma unroll
  for (int k = 0; k < KCONV; ++k) {
    // circular shift: row (l + k) mod 256
    const int r0 = ((wv << 4) + m + k) & (LSEQ - 1);            // M-tile wv
    const int r1 = (128 + (wv << 4) + m + k) & (LSEQ - 1);      // M-tile wv+8
    const unsigned short* arow0 = dataL + r0 * DSTR;
    const unsigned short* arow1 = dataL + r1 * DSTR;
#pragma unroll
    for (int kc = 0; kc < 4; ++kc) {           // K chunks of 32 within HID
      ABReg a0, a1;
      const int h0 = kc * 32 + hi * 8;         // ISA 16-bit A layout: K = hi*8+e | 16+hi*8+e
      a0.q[0] = *(const uint4*)(arow0 + h0);
      a0.q[1] = *(const uint4*)(arow0 + h0 + 16);
      a1.q[0] = *(const uint4*)(arow1 + h0);
      a1.q[1] = *(const uint4*)(arow1 + h0 + 16);
      const int hb = kc * 32 + hi * 8;         // B layout matches A: K = hi*8+e | 16+hi*8+e
#pragma unroll
      for (int nt = 0; nt < 8; ++nt) {         // 8 N-tiles cover 128 output cols
        ABReg b;
        const unsigned short* brow = wL + (k * HID + nt * 16 + m) * DSTR + hb;
        b.q[0] = *(const uint4*)(brow);
        b.q[1] = *(const uint4*)(brow + 16);
        acc[0][nt] = __builtin_amdgcn_wmma_f32_16x16x32_bf16(
            false, a0.v, false, b.v, (short)0, acc[0][nt], false, false);
        acc[1][nt] = __builtin_amdgcn_wmma_f32_16x16x32_bf16(
            false, a1.v, false, b.v, (short)0, acc[1][nt], false, false);
      }
    }
  }

  // ---- elu + reduce over rows into zbar (LDS float atomics) ----
#pragma unroll
  for (int mt = 0; mt < MT; ++mt) {
#pragma unroll
    for (int nt = 0; nt < 8; ++nt) {
      float s = 0.f;
#pragma unroll
      for (int r = 0; r < 8; ++r) s += eluf(acc[mt][nt][r]);  // rows m = hi*8 + r
      atomicAdd(&zbarL[nt * 16 + m], s);                      // column n = lane&15
    }
  }
  __syncthreads();

  // ---- z2 = elu(W_z_final @ zbar) : tiny matvec, 128 threads ----
  if (tid < HID) {
    const float* wrow = wz_final + tid * HID;
    float s = 0.f;
#pragma unroll 8
    for (int h = 0; h < HID; ++h) s += wrow[h] * zbarL[h];
    z2out[bc * HID + tid] = eluf(s);
  }
}

// out[b] = W_p_final @ (W_p @ sum_c z2[b,c])
__global__ __launch_bounds__(128) void ptree_p_kernel(
    const float* __restrict__ z2, const float* __restrict__ wp,
    const float* __restrict__ wpf, float* __restrict__ out) {
  __shared__ float zs[HID];
  __shared__ float ps[HID];
  const int b = blockIdx.x, t = threadIdx.x;
  float s = 0.f;
#pragma unroll
  for (int c = 0; c < CC; ++c) s += z2[(b * CC + c) * HID + t];
  zs[t] = s;
  __syncthreads();
  float p = 0.f;
#pragma unroll 8
  for (int h = 0; h < HID; ++h) p += wp[t * HID + h] * zs[h];
  ps[t] = p;
  __syncthreads();
  float o = 0.f;
#pragma unroll 8
  for (int h = 0; h < HID; ++h) o += wpf[t * HID + h] * ps[h];
  out[b * HID + t] = o;
}

extern "C" void kernel_launch(void* const* d_in, const int* in_sizes, int n_in,
                              void* d_out, int out_size, void* d_ws, size_t ws_size,
                              hipStream_t stream) {
  const float* x          = (const float*)d_in[0];
  const int*   leaf_idx   = (const int*)  d_in[1];
  const float* W_z        = (const float*)d_in[2];
  const float* W_z_final  = (const float*)d_in[3];
  const float* W_p        = (const float*)d_in[4];
  const float* W_p_final  = (const float*)d_in[5];
  float* out = (float*)d_out;

  const int n_x  = in_sizes[0];   // N_NODES*HID
  const int n_wz = in_sizes[2];   // K*HID*HID

  // workspace layout (256B aligned): x_bf16 | W_z_bf16 | z2 f32
  unsigned short* xbf = (unsigned short*)d_ws;
  size_t off = ((size_t)n_x * 2 + 255) & ~(size_t)255;
  unsigned short* wzbf = (unsigned short*)((char*)d_ws + off);
  off += ((size_t)n_wz * 2 + 255) & ~(size_t)255;
  float* z2 = (float*)((char*)d_ws + off);

  cvt_bf16_kernel<<<(n_x + 255) / 256, 256, 0, stream>>>(x, xbf, n_x);
  cvt_bf16_kernel<<<(n_wz + 255) / 256, 256, 0, stream>>>(W_z, wzbf, n_wz);

  const size_t smem = (size_t)(LSEQ * DSTR + KCONV * HID * DSTR) * 2 + HID * 4; // ~209.4KB < 320KB WGP LDS
  ptree_z_kernel<<<BB * CC, TPB, smem, stream>>>(xbf, leaf_idx, wzbf, W_z_final, z2);
  ptree_p_kernel<<<BB, 128, 0, stream>>>(z2, W_p, W_p_final, out);
}
